// Seq2SeqLSTM_17944373363317
// MI455X (gfx1250) — compile-verified, matrix operand order checked by reading, not run on hardware
//
#include <hip/hip_runtime.h>

typedef __attribute__((ext_vector_type(16))) __bf16 v16bf;
typedef __attribute__((ext_vector_type(8)))  float  v8f;
typedef __attribute__((ext_vector_type(4)))  int    v4i;

#define LDSTR 40                       // padded LDS row stride in halves (80B, 16B-aligned)
#define STAGE ((64 + 128) * LDSTR)     // halves per double-buffer stage

#if __has_builtin(__builtin_amdgcn_global_load_async_to_lds_b128)
#define HAVE_ASYNC_LDS 1
#endif

union Frag { uint4 q[2]; v16bf v; };

__device__ __forceinline__ unsigned short f2bf(float f) {
    unsigned u = __float_as_uint(f);
    u += 0x7FFFu + ((u >> 16) & 1u);          // round-to-nearest-even
    return (unsigned short)(u >> 16);
}
__device__ __forceinline__ float sigf(float x) { return 1.0f / (1.0f + __expf(-x)); }

// async 16B global -> LDS copy (DMA path, ASYNCcnt); sync fallback
__device__ __forceinline__ void async_cp16(const unsigned short* g, unsigned short* l) {
#if defined(HAVE_ASYNC_LDS)
    __builtin_amdgcn_global_load_async_to_lds_b128(
        (__attribute__((address_space(1))) v4i*)(void*)const_cast<unsigned short*>(g),
        (__attribute__((address_space(3))) v4i*)(void*)l, 0, 0);
#else
    *(uint4*)l = *(const uint4*)g;
#endif
}

__device__ __forceinline__ void wait_async_le(int n) {
#if defined(HAVE_ASYNC_LDS)
#if __has_builtin(__builtin_amdgcn_s_wait_asynccnt)
    if (n == 0) __builtin_amdgcn_s_wait_asynccnt(0);
    else        __builtin_amdgcn_s_wait_asynccnt(3);
#else
    if (n == 0) asm volatile("s_wait_asynccnt 0x0" ::: "memory");
    else        asm volatile("s_wait_asynccnt 0x3" ::: "memory");
#endif
#else
    (void)n;
#endif
}

// issue one stage of async staging: A tile 64x32, B tile 128x32 (3 async ops/wave)
__device__ __forceinline__ void stage_tiles(
    const unsigned short* __restrict__ A, const unsigned short* __restrict__ B,
    int K, int mBase, int nBase, int kk, unsigned short* ldsbuf)
{
    const int tid = threadIdx.x;
    unsigned short* ldsA = ldsbuf;
    unsigned short* ldsB = ldsbuf + 64 * LDSTR;
    {   // A: 256 chunks of 16B, one per thread
        int r = tid >> 2, c4 = tid & 3;
        async_cp16(A + (size_t)(mBase + r) * K + kk + c4 * 8, ldsA + r * LDSTR + c4 * 8);
    }
    #pragma unroll
    for (int it = 0; it < 2; ++it) {   // B: 512 chunks, two per thread
        int idx = tid + it * 256;
        int r = idx >> 2, c4 = idx & 3;
        async_cp16(B + (size_t)(nBase + r) * K + kk + c4 * 8, ldsB + r * LDSTR + c4 * 8);
    }
}

// ---------------- generic bf16 WMMA GEMM mainloop ----------------
// C[M,N] += A[M,K](bf16, row-major) * B[N,K](bf16, row-major)^T
// block: 256 threads = 8 waves; block tile 64(M) x 128(N); wave tile 16 x 64.
// Double-buffered LDS fed by GLOBAL_LOAD_ASYNC_TO_LDS_B128.
__device__ __forceinline__ void gemm_mainloop(
    const unsigned short* __restrict__ A, const unsigned short* __restrict__ B,
    int K, int mBase, int nBase, unsigned short* lds, v8f acc[4])
{
    const int lane = threadIdx.x & 31;
    const int wave = threadIdx.x >> 5;
    const int wm   = wave & 3;      // 4 M-tiles of 16
    const int wn   = wave >> 2;     // 2 N-strips of 64
    const int half = lane >> 4;
    const int lc   = lane & 15;
    const int r0   = half * 8;      // K-run base per ISA 16-bit A/B layout
    const int nk   = K >> 5;

    stage_tiles(A, B, K, mBase, nBase, 0, lds);            // prologue: stage 0

    for (int ki = 0; ki < nk; ++ki) {
        const bool more = (ki + 1) < nk;
        if (more)                                           // overlap: DMA next stage
            stage_tiles(A, B, K, mBase, nBase, (ki + 1) << 5, lds + ((ki + 1) & 1) * STAGE);
        wait_async_le(more ? 3 : 0);                        // stage ki complete (per wave)
        __syncthreads();                                    // all waves' DMA visible

        unsigned short* bufA = lds + (ki & 1) * STAGE;
        unsigned short* bufB = bufA + 64 * LDSTR;

        Frag fa;                                            // A frag shared by 4 N-tiles
        {
            int arow = wm * 16 + lc;
            fa.q[0] = *(const uint4*)(bufA + arow * LDSTR + r0);
            fa.q[1] = *(const uint4*)(bufA + arow * LDSTR + r0 + 16);
        }
        #pragma unroll
        for (int nt = 0; nt < 4; ++nt) {
            Frag fb;
            int brow = wn * 64 + nt * 16 + lc;
            fb.q[0] = *(const uint4*)(bufB + brow * LDSTR + r0);
            fb.q[1] = *(const uint4*)(bufB + brow * LDSTR + r0 + 16);
            acc[nt] = __builtin_amdgcn_wmma_f32_16x16x32_bf16(
                false, fa.v, false, fb.v, (short)0, acc[nt], false, false);
        }
        __syncthreads();                                    // reads done before buf reuse
    }
}

// plain epilogue: C[M,N] row-major f32
__global__ __launch_bounds__(256) void gemm_bf16(
    const unsigned short* __restrict__ A, const unsigned short* __restrict__ B,
    float* __restrict__ C, int M, int N, int K)
{
    __shared__ __align__(16) unsigned short lds[2 * STAGE];
    const int mBase = blockIdx.y * 64;
    const int nBase = blockIdx.x * 128;
    v8f acc[4] = {};
    gemm_mainloop(A, B, K, mBase, nBase, lds, acc);

    const int lane = threadIdx.x & 31, wave = threadIdx.x >> 5;
    const int wm = wave & 3, wn = wave >> 2;
    const int half = lane >> 4, lc = lane & 15;
    #pragma unroll
    for (int nt = 0; nt < 4; ++nt) {
        #pragma unroll
        for (int j = 0; j < 8; ++j) {
            int m = mBase + wm * 16 + half * 8 + j;
            int n = nBase + wn * 64 + nt * 16 + lc;
            C[(size_t)m * N + n] = acc[nt][j];
        }
    }
}

// projection epilogue: m = t*64+b  ->  out[(b*63+t)*V + v] = acc + fc_b[v]
// non-temporal stores: 516MB streaming output must not evict fc_w from L2
__global__ __launch_bounds__(256) void gemm_bf16_proj(
    const unsigned short* __restrict__ A, const unsigned short* __restrict__ B,
    const float* __restrict__ fcb, float* __restrict__ out, int K)
{
    __shared__ __align__(16) unsigned short lds[2 * STAGE];
    const int mBase = blockIdx.y * 64;
    const int nBase = blockIdx.x * 128;
    v8f acc[4] = {};
    gemm_mainloop(A, B, K, mBase, nBase, lds, acc);

    const int lane = threadIdx.x & 31, wave = threadIdx.x >> 5;
    const int wm = wave & 3, wn = wave >> 2;
    const int half = lane >> 4, lc = lane & 15;
    #pragma unroll
    for (int nt = 0; nt < 4; ++nt) {
        #pragma unroll
        for (int j = 0; j < 8; ++j) {
            int m = mBase + wm * 16 + half * 8 + j;   // m = t*64 + b
            int t = m >> 6, b = m & 63;
            int v = nBase + wn * 64 + nt * 16 + lc;
            __builtin_nontemporal_store(acc[nt][j] + fcb[v],
                                        out + ((size_t)b * 63 + t) * 32000 + v);
        }
    }
}

// ---------------- elementwise kernels ----------------

__global__ void cvt_f32_bf16(const float* __restrict__ in, unsigned short* __restrict__ out, int n) {
    int i = blockIdx.x * blockDim.x + threadIdx.x;
    if (i < n) out[i] = f2bf(__builtin_nontemporal_load(in + i));
}

// gather encoder embeddings: Xe[(t*64+b)*256 + e] = emb[src[b*128+t]*256 + e]
__global__ void embed_enc(const int* __restrict__ src, const float* __restrict__ emb,
                          unsigned short* __restrict__ Xe) {
    int i = blockIdx.x * blockDim.x + threadIdx.x;     // over 8192*256
    int row = i >> 8, e = i & 255;
    int t = row >> 6, b = row & 63;
    int tok = src[b * 128 + t];
    Xe[i] = f2bf(emb[(size_t)tok * 256 + e]);
}

// gather decoder embeddings (tgt[:, :-1]): Xd[(t*64+b)*256 + e] = emb[tgt[b*64+t]*256 + e]
__global__ void embed_dec(const int* __restrict__ tgt, const float* __restrict__ emb,
                          unsigned short* __restrict__ Xd) {
    int i = blockIdx.x * blockDim.x + threadIdx.x;     // over 4032*256
    int row = i >> 8, e = i & 255;
    int t = row >> 6, b = row & 63;
    int tok = tgt[b * 64 + t];
    Xd[i] = f2bf(emb[(size_t)tok * 256 + e]);
}

__global__ void init_state(float* h32, float* c32, unsigned short* hbf) {
    int i = blockIdx.x * blockDim.x + threadIdx.x;     // over 64*512
    h32[i] = 0.0f; c32[i] = 0.0f; hbf[i] = 0;
}

// LSTM gate math.  Gtmp = h@Whh^T for this step; Gpre_t = precomputed x@Wih^T.
// Encoder: src_lens != null -> masked carry.  Decoder: HsRow != null -> write hs tile.
__global__ void lstm_cell(const float* __restrict__ Gtmp, const float* __restrict__ Gpre_t,
                          const float* __restrict__ bias,
                          float* __restrict__ h32, float* __restrict__ c32,
                          unsigned short* __restrict__ hbf,
                          unsigned short* __restrict__ HsRow,
                          const int* __restrict__ src_lens, int t)
{
    int i = blockIdx.x * blockDim.x + threadIdx.x;     // over 64*512
    int b = i >> 9, j = i & 511;
    size_t base = (size_t)b * 2048 + j;
    float gi = Gtmp[base]        + Gpre_t[base]        + bias[j];
    float gf = Gtmp[base + 512]  + Gpre_t[base + 512]  + bias[j + 512];
    float gg = Gtmp[base + 1024] + Gpre_t[base + 1024] + bias[j + 1024];
    float go = Gtmp[base + 1536] + Gpre_t[base + 1536] + bias[j + 1536];

    float c_old = c32[i];
    float cn = sigf(gf) * c_old + sigf(gi) * tanhf(gg);
    float hn = sigf(go) * tanhf(cn);

    float h_new = hn, c_new = cn;
    if (src_lens) {                       // encoder: masked carry
        bool m = t < src_lens[b];
        h_new = m ? hn : h32[i];
        c_new = m ? cn : c_old;
    }
    h32[i] = h_new;
    c32[i] = c_new;
    hbf[i] = f2bf(h_new);
    if (HsRow) HsRow[(size_t)b * 512 + j] = f2bf(hn);
}

// ---------------- host launch ----------------
extern "C" void kernel_launch(void* const* d_in, const int* in_sizes, int n_in,
                              void* d_out, int out_size, void* d_ws, size_t ws_size,
                              hipStream_t stream) {
    (void)in_sizes; (void)n_in; (void)out_size; (void)ws_size;
    const int*   src      = (const int*)  d_in[0];
    const int*   src_lens = (const int*)  d_in[1];
    const int*   tgt      = (const int*)  d_in[2];
    const float* emb_enc  = (const float*)d_in[3];
    const float* Wih_e    = (const float*)d_in[4];
    const float* Whh_e    = (const float*)d_in[5];
    const float* b_e      = (const float*)d_in[6];
    const float* emb_dec  = (const float*)d_in[7];
    const float* Wih_d    = (const float*)d_in[8];
    const float* Whh_d    = (const float*)d_in[9];
    const float* b_d      = (const float*)d_in[10];
    const float* fc_w     = (const float*)d_in[11];
    const float* fc_b     = (const float*)d_in[12];

    char* ws = (char*)d_ws;
    size_t off = 0;
    auto alloc = [&](size_t bytes) -> void* {
        void* p = ws + off;
        off += (bytes + 255) & ~(size_t)255;
        return p;
    };
    unsigned short* WihE = (unsigned short*)alloc((size_t)2048 * 256 * 2);
    unsigned short* WhhE = (unsigned short*)alloc((size_t)2048 * 512 * 2);
    unsigned short* WihD = (unsigned short*)alloc((size_t)2048 * 256 * 2);
    unsigned short* WhhD = (unsigned short*)alloc((size_t)2048 * 512 * 2);
    unsigned short* FcW  = (unsigned short*)alloc((size_t)32000 * 512 * 2);
    unsigned short* Xe   = (unsigned short*)alloc((size_t)8192 * 256 * 2);
    unsigned short* Xd   = (unsigned short*)alloc((size_t)4032 * 256 * 2);
    float* GpreE = (float*)alloc((size_t)8192 * 2048 * 4);
    float* GpreD = (float*)alloc((size_t)4032 * 2048 * 4);
    float* Gtmp  = (float*)alloc((size_t)64 * 2048 * 4);
    float* h32   = (float*)alloc((size_t)64 * 512 * 4);
    float* c32   = (float*)alloc((size_t)64 * 512 * 4);
    unsigned short* hbf = (unsigned short*)alloc((size_t)64 * 512 * 2);
    unsigned short* Hs  = (unsigned short*)alloc((size_t)4032 * 512 * 2);

    const int TB = 256;
    // 1) convert weights to bf16 (one-shot, non-temporal reads)
    cvt_f32_bf16<<<(2048 * 256 + TB - 1) / TB, TB, 0, stream>>>(Wih_e, WihE, 2048 * 256);
    cvt_f32_bf16<<<(2048 * 512 + TB - 1) / TB, TB, 0, stream>>>(Whh_e, WhhE, 2048 * 512);
    cvt_f32_bf16<<<(2048 * 256 + TB - 1) / TB, TB, 0, stream>>>(Wih_d, WihD, 2048 * 256);
    cvt_f32_bf16<<<(2048 * 512 + TB - 1) / TB, TB, 0, stream>>>(Whh_d, WhhD, 2048 * 512);
    cvt_f32_bf16<<<(32000 * 512 + TB - 1) / TB, TB, 0, stream>>>(fc_w, FcW, 32000 * 512);

    // 2) embedding gathers
    embed_enc<<<(8192 * 256) / TB, TB, 0, stream>>>(src, emb_enc, Xe);
    embed_dec<<<(4032 * 256) / TB, TB, 0, stream>>>(tgt, emb_dec, Xd);

    // 3) batched input pre-gate GEMMs (all timesteps in parallel)
    gemm_bf16<<<dim3(2048 / 128, 8192 / 64), TB, 0, stream>>>(Xe, WihE, GpreE, 8192, 2048, 256);
    gemm_bf16<<<dim3(2048 / 128, 4032 / 64), TB, 0, stream>>>(Xd, WihD, GpreD, 4032, 2048, 256);

    // 4) zero initial state
    init_state<<<(64 * 512) / TB, TB, 0, stream>>>(h32, c32, hbf);

    // 5) encoder recurrence: serial path = h@Whh^T GEMM + gate kernel per step
    for (int t = 0; t < 128; ++t) {
        gemm_bf16<<<dim3(16, 1), TB, 0, stream>>>(hbf, WhhE, Gtmp, 64, 2048, 512);
        lstm_cell<<<(64 * 512) / TB, TB, 0, stream>>>(
            Gtmp, GpreE + (size_t)t * 64 * 2048, b_e, h32, c32, hbf,
            nullptr, src_lens, t);
    }
    // 6) decoder recurrence (teacher forcing), collect hs in bf16
    for (int t = 0; t < 63; ++t) {
        gemm_bf16<<<dim3(16, 1), TB, 0, stream>>>(hbf, WhhD, Gtmp, 64, 2048, 512);
        lstm_cell<<<(64 * 512) / TB, TB, 0, stream>>>(
            Gtmp, GpreD + (size_t)t * 64 * 2048, b_d, h32, c32, hbf,
            Hs + (size_t)t * 64 * 512, nullptr, t);
    }

    // 7) output projection: [4032,512] x [32000,512]^T + fc_b -> logits [64,63,32000]
    gemm_bf16_proj<<<dim3(32000 / 128, 4032 / 64), TB, 0, stream>>>(
        Hs, FcW, fc_b, (float*)d_out, 512);
}
